// RNN_5763846111354
// MI455X (gfx1250) — matrix operand order repaired and checked by reading, not compiled, not run on hardware
//
#include <hip/hip_runtime.h>
#include <hip/hip_bf16.h>

#define TT   128
#define BB   128
#define DIN  512
#define HH   1024
#define NOUT 1000
#define G4H  (4 * HH)
#define NBLK 64   // persistent grid: block = h-col tile (64), wave = batch tile (8)

typedef __attribute__((ext_vector_type(16))) __bf16 v16bf;
typedef __attribute__((ext_vector_type(8)))  float  v8f;
typedef __attribute__((ext_vector_type(4)))  unsigned int u32x4;

union BFrag { v16bf v; u32x4 u[2]; };

// ---- fragment loaders (wave32 WMMA layouts per CDNA5 ISA 7.12.2) ----
// A 16x32 bf16: rowptr already includes the lane's row (m = lane&15).
// lanes 0-15 hold K = {k0..k0+7, k0+16..k0+23}; lanes 16-31 hold +8.
__device__ __forceinline__ v16bf load_a16x32(const __bf16* rowptr, int k0, int lane) {
    const __bf16* p = rowptr + k0 + ((lane >> 4) << 3);
    BFrag f;
    f.u[0] = *(const u32x4*)(p);
    f.u[1] = *(const u32x4*)(p + 16);
    return f.v;
}
// B 32x16 bf16: rowptr includes the lane's column row of W (n = lane&15);
// lanes 0-15 hold K = {k0..k0+7, k0+16..k0+23}; lanes 16-31 hold +8.
__device__ __forceinline__ v16bf load_b32x16(const __bf16* rowptr, int k0, int lane) {
    const __bf16* p = rowptr + k0 + ((lane >> 4) << 3);
    BFrag f;
    f.u[0] = *(const u32x4*)(p);
    f.u[1] = *(const u32x4*)(p + 16);
    return f.v;
}

__device__ __forceinline__ v8f wmma_bf16(v16bf a, v16bf b, v8f c) {
    return __builtin_amdgcn_wmma_f32_16x16x32_bf16(false, a, false, b, (short)0, c,
                                                   false, false);
}

__device__ __forceinline__ float fsig(float x) {
    return 1.0f / (1.0f + __expf(-x));
}
__device__ __forceinline__ float ftanh(float x) {
    x = fminf(fmaxf(x, -15.0f), 15.0f);
    float e = __expf(2.0f * x);
    return (e - 1.0f) / (e + 1.0f);
}

// ---- small prep kernels ----
__global__ void cast_f32_bf16(const float* __restrict__ in, __bf16* __restrict__ out, size_t n) {
    for (size_t i = (size_t)blockIdx.x * blockDim.x + threadIdx.x; i < n;
         i += (size_t)gridDim.x * blockDim.x)
        out[i] = (__bf16)in[i];
}

// x [B,T,D] fp32 -> xb [T,B,D] bf16
__global__ void transpose_cast_x(const float* __restrict__ x, __bf16* __restrict__ xb) {
    size_t i = (size_t)blockIdx.x * blockDim.x + threadIdx.x;
    if (i >= (size_t)TT * BB * DIN) return;
    int    d = (int)(i % DIN);
    size_t r = i / DIN;
    int    b = (int)(r % BB);
    int    t = (int)(r / BB);
    xb[i] = (__bf16)x[((size_t)b * TT + t) * DIN + d];
}

__global__ void add_bias_vec(const float* __restrict__ a, const float* __restrict__ b,
                             float* __restrict__ o, int n) {
    int i = blockIdx.x * blockDim.x + threadIdx.x;
    if (i < n) o[i] = a[i] + b[i];
}

// ---- WMMA GEMM, 2x2 register blocking: each wave computes a 32x32 tile ----
// out[M,ldo] = A[M,K] @ W[Nreal,K]^T + bias
__global__ __launch_bounds__(256) void wmma_gemm_bias(
    const __bf16* __restrict__ A, int K,
    const __bf16* __restrict__ W,
    const float* __restrict__ bias,
    float* __restrict__ out, int ldo,
    int Mt2, int Nt2, int Nreal)
{
    const int g = blockIdx.x * 8 + (threadIdx.x >> 5);
    if (g >= Mt2 * Nt2) return;  // whole-wave uniform exit -> EXEC all-1 for WMMA
    const int mt = g / Nt2;
    const int nt = g % Nt2;
    const int lane = threadIdx.x & 31;
    const int lc = lane & 15;

    const int ncol0 = nt * 32 + lc;
    const int ncol1 = ncol0 + 16;
    const int nr0 = (ncol0 < Nreal) ? ncol0 : (Nreal - 1);
    const int nr1 = (ncol1 < Nreal) ? ncol1 : (Nreal - 1);

    const __bf16* arow0 = A + (size_t)(mt * 32 + lc) * K;
    const __bf16* arow1 = arow0 + (size_t)16 * K;
    const __bf16* wrow0 = W + (size_t)nr0 * K;
    const __bf16* wrow1 = W + (size_t)nr1 * K;

    v8f a00 = {}, a01 = {}, a10 = {}, a11 = {};
    for (int k0 = 0; k0 < K; k0 += 32) {
        v16bf A0 = load_a16x32(arow0, k0, lane);
        v16bf A1 = load_a16x32(arow1, k0, lane);
        v16bf B0 = load_b32x16(wrow0, k0, lane);
        v16bf B1 = load_b32x16(wrow1, k0, lane);
        a00 = wmma_bf16(A0, B0, a00);
        a01 = wmma_bf16(A0, B1, a01);
        a10 = wmma_bf16(A1, B0, a10);
        a11 = wmma_bf16(A1, B1, a11);
    }

    const int m0 = mt * 32 + ((lane >> 4) << 3);
    if (ncol0 < Nreal) {
        const float bv = bias[ncol0];
#pragma unroll
        for (int j = 0; j < 8; ++j) {
            out[(size_t)(m0 + j) * ldo + ncol0]      = a00[j] + bv;
            out[(size_t)(m0 + 16 + j) * ldo + ncol0] = a10[j] + bv;
        }
    }
    if (ncol1 < Nreal) {
        const float bv = bias[ncol1];
#pragma unroll
        for (int j = 0; j < 8; ++j) {
            out[(size_t)(m0 + j) * ldo + ncol1]      = a01[j] + bv;
            out[(size_t)(m0 + 16 + j) * ldo + ncol1] = a11[j] + bv;
        }
    }
}

// ---- device-wide split barrier (persistent grid of NBLK blocks) ----
__device__ __forceinline__ void grid_barrier(unsigned* cnt, unsigned* phase) {
    __syncthreads();
    if (threadIdx.x == 0) {
        __builtin_amdgcn_fence(__ATOMIC_RELEASE, "agent");
        __hip_atomic_fetch_add(cnt, 1u, __ATOMIC_RELAXED, __HIP_MEMORY_SCOPE_AGENT);
        *phase += NBLK;
        while (__hip_atomic_load(cnt, __ATOMIC_RELAXED, __HIP_MEMORY_SCOPE_AGENT) < *phase)
            __builtin_amdgcn_s_sleep(8);
    }
    __syncthreads();
    __builtin_amdgcn_fence(__ATOMIC_ACQUIRE, "agent");
}

// ---- persistent LSTM layer ----
// block = h-column tile (n16 = blockIdx*16), wave = batch tile.
// The block's W_hh slice (4 gates x 16 cols x HH, 128KB bf16) is staged into
// LDS ONCE and reused by all 8 waves for all T steps; only the h tile and the
// precomputed xg addend touch global memory per step. Cell state c stays in
// registers for the whole sequence; h is double-buffered by step parity so a
// single device-wide barrier per step is sufficient.
#define LDW (HH + 8)                       // padded LDS row stride (bank rotate)
#define LSTM_LDS_BYTES (64 * LDW * 2)      // 64 rows x (HH+8) bf16 = 129 KB

__global__ __launch_bounds__(256, 1) void lstm_layer(
    const float* __restrict__ xg,    // [T*B, 4H] fp32, biases folded in
    const __bf16* __restrict__ Whh,  // [4H, H] bf16
    __bf16* __restrict__ hstate,     // [2, B, H] bf16, parity-0 buffer pre-zeroed
    __bf16* __restrict__ hseq,       // [T*B, H] bf16 or nullptr
    unsigned* __restrict__ cnt)
{
    const int lane = threadIdx.x & 31;
    const int bt   = threadIdx.x >> 5;   // batch tile = wave id (0..7)
    const int n16  = blockIdx.x << 4;    // h-col tile = block id (0..63)
    const int nc   = n16 + (lane & 15);
    const int mhi  = (lane >> 4) << 3;

    extern __shared__ __align__(16) char smem_raw[];
    __bf16* lw = (__bf16*)smem_raw;

    // stage W_hh slice: LDS row r (= gate*16 + col) <- W_hh[gate*H + n16 + col]
    for (int idx = threadIdx.x; idx < 64 * (HH / 8); idx += 256) {
        const int r  = idx >> 7;          // 0..63
        const int k8 = idx & 127;         // 8-elem chunk within row
        const int q  = r >> 4;
        const int c  = r & 15;
        *(u32x4*)(lw + (size_t)r * LDW + k8 * 8) =
            *(const u32x4*)(Whh + (size_t)(q * HH + n16 + c) * HH + k8 * 8);
    }
    __syncthreads();

    const __bf16* l_i = lw + (size_t)(0 * 16 + (lane & 15)) * LDW;
    const __bf16* l_f = lw + (size_t)(1 * 16 + (lane & 15)) * LDW;
    const __bf16* l_g = lw + (size_t)(2 * 16 + (lane & 15)) * LDW;
    const __bf16* l_o = lw + (size_t)(3 * 16 + (lane & 15)) * LDW;

    v8f creg = {};           // cell state lives in registers for all T steps
    unsigned phase = 0;

    for (int t = 0; t < TT; ++t) {
        const __bf16* hrow = hstate + (size_t)(t & 1) * (BB * HH)
                                    + (size_t)(bt * 16 + (lane & 15)) * HH;
        v8f ai = {}, af = {}, ag = {}, ao = {};
#pragma unroll 4
        for (int k0 = 0; k0 < HH; k0 += 32) {
            v16bf a = load_a16x32(hrow, k0, lane);
            ai = wmma_bf16(a, load_b32x16(l_i, k0, lane), ai);
            af = wmma_bf16(a, load_b32x16(l_f, k0, lane), af);
            ag = wmma_bf16(a, load_b32x16(l_g, k0, lane), ag);
            ao = wmma_bf16(a, load_b32x16(l_o, k0, lane), ao);
        }

        const size_t xgrow = ((size_t)t * BB + bt * 16 + mhi) * G4H;
        __bf16* hw = hstate + (size_t)(1 - (t & 1)) * (BB * HH);
#pragma unroll
        for (int j = 0; j < 8; ++j) {
            const size_t xr = xgrow + (size_t)j * G4H;
            float iv = fsig (ai[j] + xg[xr + 0 * HH + nc]);
            float fv = fsig (af[j] + xg[xr + 1 * HH + nc]);
            float gv = ftanh(ag[j] + xg[xr + 2 * HH + nc]);
            float ov = fsig (ao[j] + xg[xr + 3 * HH + nc]);
            float cv = fv * creg[j] + iv * gv;
            creg[j]  = cv;
            float hv = ov * ftanh(cv);
            __bf16 hb = (__bf16)hv;
            const int brow = bt * 16 + mhi + j;
            hw[(size_t)brow * HH + nc] = hb;
            if (hseq) hseq[((size_t)t * BB + brow) * HH + nc] = hb;
        }
        grid_barrier(cnt, &phase);  // double-buffered h => one barrier/step is safe
    }
}

extern "C" void kernel_launch(void* const* d_in, const int* in_sizes, int n_in,
                              void* d_out, int out_size, void* d_ws, size_t ws_size,
                              hipStream_t stream) {
    const float* x     = (const float*)d_in[0];
    const float* Wih0  = (const float*)d_in[1];
    const float* Whh0  = (const float*)d_in[2];
    const float* bih0  = (const float*)d_in[3];
    const float* bhh0  = (const float*)d_in[4];
    const float* Wih1  = (const float*)d_in[5];
    const float* Whh1  = (const float*)d_in[6];
    const float* bih1  = (const float*)d_in[7];
    const float* bhh1  = (const float*)d_in[8];
    const float* Wfc   = (const float*)d_in[9];
    const float* bfc   = (const float*)d_in[10];
    float* out = (float*)d_out;

    char* ws = (char*)d_ws;
    size_t off = 0;
    auto alloc = [&](size_t bytes) -> void* {
        off = (off + 255) & ~(size_t)255;
        void* p = ws + off;
        off += bytes;
        return p;
    };

    unsigned* cnt  = (unsigned*)alloc(256);
    float* b0      = (float*)alloc((size_t)G4H * 4);
    float* b1      = (float*)alloc((size_t)G4H * 4);
    __bf16* hst0   = (__bf16*)alloc((size_t)2 * BB * HH * 2);
    __bf16* hst1   = (__bf16*)alloc((size_t)2 * BB * HH * 2);
    __bf16* xb     = (__bf16*)alloc((size_t)TT * BB * DIN * 2);
    __bf16* wih0b  = (__bf16*)alloc((size_t)G4H * DIN * 2);
    __bf16* whh0b  = (__bf16*)alloc((size_t)G4H * HH * 2);
    __bf16* wih1b  = (__bf16*)alloc((size_t)G4H * HH * 2);
    __bf16* whh1b  = (__bf16*)alloc((size_t)G4H * HH * 2);
    __bf16* wfcb   = (__bf16*)alloc((size_t)NOUT * HH * 2);
    __bf16* h1buf  = (__bf16*)alloc((size_t)TT * BB * HH * 2);
    float*  xgbuf  = (float*)alloc((size_t)TT * BB * G4H * 4);

    hipMemsetAsync(cnt, 0, 256, stream);
    hipMemsetAsync(hst0, 0, (size_t)2 * BB * HH * 2, stream);
    hipMemsetAsync(hst1, 0, (size_t)2 * BB * HH * 2, stream);

    // --- prep: transpose/cast x, cast weights, fold biases ---
    {
        size_t n = (size_t)TT * BB * DIN;
        transpose_cast_x<<<(unsigned)((n + 255) / 256), 256, 0, stream>>>(x, xb);
    }
    cast_f32_bf16<<<2048, 256, 0, stream>>>(Wih0, wih0b, (size_t)G4H * DIN);
    cast_f32_bf16<<<2048, 256, 0, stream>>>(Whh0, whh0b, (size_t)G4H * HH);
    cast_f32_bf16<<<2048, 256, 0, stream>>>(Wih1, wih1b, (size_t)G4H * HH);
    cast_f32_bf16<<<2048, 256, 0, stream>>>(Whh1, whh1b, (size_t)G4H * HH);
    cast_f32_bf16<<<2048, 256, 0, stream>>>(Wfc,  wfcb,  (size_t)NOUT * HH);
    add_bias_vec<<<(G4H + 255) / 256, 256, 0, stream>>>(bih0, bhh0, b0, G4H);
    add_bias_vec<<<(G4H + 255) / 256, 256, 0, stream>>>(bih1, bhh1, b1, G4H);

    // --- layer 0: xg0 = xb @ Wih0^T + b0 ; then persistent recurrence ---
    {
        int mt2 = (TT * BB) / 32, nt2 = (G4H / 16 + 1) / 2;
        int tiles = mt2 * nt2;
        wmma_gemm_bias<<<(tiles + 7) / 8, 256, 0, stream>>>(
            xb, DIN, wih0b, b0, xgbuf, G4H, mt2, nt2, G4H);
    }
    lstm_layer<<<NBLK, 256, LSTM_LDS_BYTES, stream>>>(xgbuf, whh0b, hst0, h1buf, cnt + 0);

    // --- layer 1: xg1 = h1 @ Wih1^T + b1 ; recurrence (no sequence output) ---
    {
        int mt2 = (TT * BB) / 32, nt2 = (G4H / 16 + 1) / 2;
        int tiles = mt2 * nt2;
        wmma_gemm_bias<<<(tiles + 7) / 8, 256, 0, stream>>>(
            h1buf, HH, wih1b, b1, xgbuf, G4H, mt2, nt2, G4H);
    }
    lstm_layer<<<NBLK, 256, LSTM_LDS_BYTES, stream>>>(xgbuf, whh1b, hst1, nullptr, cnt + 1);

    // --- FC: final h lives in hst1 parity-0 buffer (write parity of t=127 is 0) ---
    {
        int mt2 = BB / 32, nt2 = ((NOUT + 15) / 16 + 1) / 2;
        int tiles = mt2 * nt2;
        wmma_gemm_bias<<<(tiles + 7) / 8, 256, 0, stream>>>(
            hst1, HH, wfcb, bfc, out, NOUT, mt2, nt2, NOUT);
    }
}
